// FairnessBranch_37718402793541
// MI455X (gfx1250) — matrix operand order repaired and checked by reading, not compiled, not run on hardware
//
#include <hip/hip_runtime.h>
#include <hip/hip_bf16.h>
#include <math.h>

// Problem constants (match reference)
#define BDIM 65536
#define DDIM 1024
#define EDIM 16
#define H1   128
#define H2   64
#define R1   64
#define R2   32
#define EPSV 1e-5f

typedef __attribute__((ext_vector_type(16))) __bf16 v16bf;
typedef __attribute__((ext_vector_type(8)))  float  v8f;

union ABf { v16bf v; uint4 q[2]; };

static __device__ __forceinline__ void loadFrag(ABf& f, const __bf16* p) {
  f.q[0] = *(const uint4*)p;
  f.q[1] = *(const uint4*)(p + 16);
}
static __device__ __forceinline__ void loadFragA(ABf& f, const __bf16* p) {
  f.q[0] = *(const uint4*)p;
  f.q[1] = *(const uint4*)(p + 16);
}

// float -> bf16 round-to-nearest-even (bit trick; avoids relying on cast support)
static __device__ __forceinline__ __bf16 f2bf(float f) {
  union { float f; unsigned u; } a; a.f = f;
  unsigned r = a.u + 0x7FFFu + ((a.u >> 16) & 1u);
  union { unsigned short s; __bf16 b; } o; o.s = (unsigned short)(r >> 16);
  return o.b;
}

// ---------------- Prep kernels ----------------

// x fp32 -> bf16, 8 elements/thread
__global__ __launch_bounds__(256) void k_cvt_x(const float* __restrict__ x,
                                               __bf16* __restrict__ xb) {
  size_t i = ((size_t)blockIdx.x * blockDim.x + threadIdx.x) * 8;
  float4 a = *(const float4*)(x + i);
  float4 b = *(const float4*)(x + i + 4);
  union { __bf16 h[8]; uint4 q; } o;
  o.h[0] = f2bf(a.x); o.h[1] = f2bf(a.y); o.h[2] = f2bf(a.z); o.h[3] = f2bf(a.w);
  o.h[4] = f2bf(b.x); o.h[5] = f2bf(b.y); o.h[6] = f2bf(b.z); o.h[7] = f2bf(b.w);
  *(uint4*)(xb + i) = o.q;
}

// W1t[e][h][d] = eW1[e][d][h] * scale[e][d]  (BN scale folded, transposed for B-fragments)
__global__ __launch_bounds__(256) void k_fold_w1(const float* __restrict__ eW1,
                                                 const float* __restrict__ g,
                                                 const float* __restrict__ v,
                                                 __bf16* __restrict__ w1t) {
  size_t o = (size_t)blockIdx.x * blockDim.x + threadIdx.x; // e*H1*D + h*D + d
  int d = (int)(o % DDIM);
  size_t t = o / DDIM;
  int h = (int)(t % H1);
  int e = (int)(t / H1);
  float sc = g[e * DDIM + d] * rsqrtf(v[e * DDIM + d] + EPSV);
  w1t[o] = f2bf(eW1[((size_t)e * DDIM + d) * H1 + h] * sc);
}

// rW1t[r][d] = rW1[d][r] * rscale[d]
__global__ __launch_bounds__(256) void k_fold_rw1(const float* __restrict__ rW1,
                                                  const float* __restrict__ g,
                                                  const float* __restrict__ v,
                                                  __bf16* __restrict__ rw1t) {
  size_t o = (size_t)blockIdx.x * blockDim.x + threadIdx.x; // r*D + d
  int d = (int)(o % DDIM);
  int r = (int)(o / DDIM);
  float sc = g[d] * rsqrtf(v[d] + EPSV);
  rw1t[o] = f2bf(rW1[(size_t)d * R1 + r] * sc);
}

// b1_eff[e][h] = eb1[e][h] + sum_d shift[e][d] * eW1[e][d][h]
__global__ __launch_bounds__(256) void k_b1eff(const float* __restrict__ eW1,
                                               const float* __restrict__ g,
                                               const float* __restrict__ b,
                                               const float* __restrict__ m,
                                               const float* __restrict__ v,
                                               const float* __restrict__ eb1,
                                               float* __restrict__ b1e) {
  int e = blockIdx.x / H1, h = blockIdx.x % H1;
  float p = 0.f;
  for (int d = threadIdx.x; d < DDIM; d += 256) {
    float sc = g[e * DDIM + d] * rsqrtf(v[e * DDIM + d] + EPSV);
    float sh = b[e * DDIM + d] - m[e * DDIM + d] * sc;
    p += sh * eW1[((size_t)e * DDIM + d) * H1 + h];
  }
  __shared__ float red[256];
  red[threadIdx.x] = p;
  __syncthreads();
  for (int s = 128; s > 0; s >>= 1) {
    if ((int)threadIdx.x < s) red[threadIdx.x] += red[threadIdx.x + s];
    __syncthreads();
  }
  if (threadIdx.x == 0) b1e[blockIdx.x] = eb1[blockIdx.x] + red[0];
}

// rb1_eff[r] = rb1[r] + sum_d rshift[d] * rW1[d][r]
__global__ __launch_bounds__(256) void k_rb1eff(const float* __restrict__ rW1,
                                                const float* __restrict__ g,
                                                const float* __restrict__ b,
                                                const float* __restrict__ m,
                                                const float* __restrict__ v,
                                                const float* __restrict__ rb1,
                                                float* __restrict__ rb1e) {
  int r = blockIdx.x;
  float p = 0.f;
  for (int d = threadIdx.x; d < DDIM; d += 256) {
    float sc = g[d] * rsqrtf(v[d] + EPSV);
    float sh = b[d] - m[d] * sc;
    p += sh * rW1[(size_t)d * R1 + r];
  }
  __shared__ float red[256];
  red[threadIdx.x] = p;
  __syncthreads();
  for (int s = 128; s > 0; s >>= 1) {
    if ((int)threadIdx.x < s) red[threadIdx.x] += red[threadIdx.x + s];
    __syncthreads();
  }
  if (threadIdx.x == 0) rb1e[r] = rb1[r] + red[0];
}

// W2t[e][n][k] = eW2[e][k][n]
__global__ __launch_bounds__(256) void k_w2t(const float* __restrict__ eW2,
                                             __bf16* __restrict__ w2t) {
  size_t o = (size_t)blockIdx.x * blockDim.x + threadIdx.x; // e*H2*H1 + n*H1 + k
  int k = (int)(o % H1);
  size_t t = o / H1;
  int n = (int)(t % H2);
  int e = (int)(t / H2);
  w2t[o] = f2bf(eW2[((size_t)e * H1 + k) * H2 + n]);
}

// ---------------- Router: BN-folded D->64 via WMMA, then scalar 64->32->16->softmax ----------------

__global__ __launch_bounds__(256) void k_router(const __bf16* __restrict__ xb,
                                                const __bf16* __restrict__ rw1t,
                                                const float* __restrict__ rb1e,
                                                const float* __restrict__ rW2,
                                                const float* __restrict__ rb2,
                                                const float* __restrict__ rW3,
                                                const float* __restrict__ rb3,
                                                float* __restrict__ rw_out) {
  extern __shared__ __align__(16) char smem[];
  __bf16* w1s = (__bf16*)smem;                       // [64][1024] bf16 = 128KB
  float* fb   = (float*)(smem + R1 * DDIM * 2);
  float* rb1s = fb;                                  // 64
  float* w2s  = rb1s + R1;                           // 64*32 (row-major [k][j])
  float* b2s  = w2s + R1 * R2;                       // 32
  float* w3s  = b2s + R2;                            // 32*16 (row-major [j][i])
  float* b3s  = w3s + R2 * EDIM;                     // 16
  float* h1a  = b3s + EDIM;                          // 8 waves * 16 * 65 (padded)

  // cooperative staging
  for (int i = threadIdx.x; i < R1 * DDIM / 8; i += 256)
    ((uint4*)w1s)[i] = ((const uint4*)rw1t)[i];
  for (int i = threadIdx.x; i < R1; i += 256)        rb1s[i] = rb1e[i];
  for (int i = threadIdx.x; i < R1 * R2; i += 256)   w2s[i] = rW2[i];
  for (int i = threadIdx.x; i < R2; i += 256)        b2s[i] = rb2[i];
  for (int i = threadIdx.x; i < R2 * EDIM; i += 256) w3s[i] = rW3[i];
  for (int i = threadIdx.x; i < EDIM; i += 256)      b3s[i] = rb3[i];
  __syncthreads();

  const int lane = threadIdx.x & 31, w = threadIdx.x >> 5;
  const int lid = lane & 15, hi = lane >> 4;
  const int gRow = blockIdx.x * 128 + w * 16;
  const __bf16* xrow = xb + (size_t)(gRow + lid) * DDIM + hi * 8;

  // two-group software pipeline: group0 = nt{0,1}, group1 = nt{2,3}
  v8f acc[4] = {};
  ABf a0, a1, bA[2], bB[2];
  loadFragA(a0, xrow);
#pragma unroll
  for (int nt = 0; nt < 2; ++nt)
    loadFrag(bA[nt], w1s + (size_t)(nt * 16 + lid) * DDIM + hi * 8);
#pragma unroll 4
  for (int k = 0; k < 32; ++k) {
    // issue group1 loads for this step (covered by group0 WMMAs)
#pragma unroll
    for (int nt = 0; nt < 2; ++nt)
      loadFrag(bB[nt], w1s + (size_t)((nt + 2) * 16 + lid) * DDIM + k * 32 + hi * 8);
    if (k < 31) loadFragA(a1, xrow + (k + 1) * 32);
#pragma unroll
    for (int nt = 0; nt < 2; ++nt)
      acc[nt] = __builtin_amdgcn_wmma_f32_16x16x32_bf16(
          false, a0.v, false, bA[nt].v, (short)0, acc[nt], false, false);
    if (k < 31) {
      // issue next step's group0 loads (covered by group1 WMMAs)
#pragma unroll
      for (int nt = 0; nt < 2; ++nt)
        loadFrag(bA[nt], w1s + (size_t)(nt * 16 + lid) * DDIM + (k + 1) * 32 + hi * 8);
    }
#pragma unroll
    for (int nt = 0; nt < 2; ++nt)
      acc[nt + 2] = __builtin_amdgcn_wmma_f32_16x16x32_bf16(
          false, a0.v, false, bB[nt].v, (short)0, acc[nt + 2], false, false);
    if (k < 31) a0 = a1;
  }

  // relu(h1+bias) -> LDS (padded stride 65 to dodge bank conflicts in scalar phase)
  float* hw = h1a + w * (16 * 65);
#pragma unroll
  for (int nt = 0; nt < 4; ++nt) {
    int n = nt * 16 + lid;
    float bn = rb1s[n];
#pragma unroll
    for (int r = 0; r < 8; ++r)
      hw[(r + hi * 8) * 65 + n] = fmaxf(acc[nt][r] + bn, 0.f);
  }
  __syncthreads();

  // scalar tail: one thread per row
  if (threadIdx.x < 128) {
    int t = (int)threadIdx.x;
    const float* hr = h1a + (t >> 4) * (16 * 65) + (t & 15) * 65;
    float h2v[R2];
#pragma unroll
    for (int j = 0; j < R2; ++j) {
      float s = b2s[j];
#pragma unroll 8
      for (int k = 0; k < R1; ++k) s = fmaf(hr[k], w2s[k * R2 + j], s);
      h2v[j] = fmaxf(s, 0.f);
    }
    float lg[EDIM];
    float mx = -1e30f;
#pragma unroll
    for (int i = 0; i < EDIM; ++i) {
      float s = b3s[i];
#pragma unroll 8
      for (int j = 0; j < R2; ++j) s = fmaf(h2v[j], w3s[j * EDIM + i], s);
      lg[i] = s;
      mx = fmaxf(mx, s);
    }
    float sum = 0.f;
#pragma unroll
    for (int i = 0; i < EDIM; ++i) { float p = __expf(lg[i] - mx); lg[i] = p; sum += p; }
    float inv = 1.f / sum;
    int row = blockIdx.x * 128 + t;
    float* o = rw_out + (size_t)row * EDIM;
#pragma unroll
    for (int i = 0; i < EDIM; ++i) o[i] = lg[i] * inv;
  }
}

// ---------------- Experts: fused D->128->64->1 -> sigmoid -> weighted sum ----------------

__global__ __launch_bounds__(256) void k_experts(const __bf16* __restrict__ xb,
                                                 const __bf16* __restrict__ w1t,
                                                 const float* __restrict__ b1e,
                                                 const __bf16* __restrict__ w2t,
                                                 const float* __restrict__ eW3,
                                                 const float* __restrict__ eb2g,
                                                 const float* __restrict__ eb3,
                                                 const float* __restrict__ rw,
                                                 float* __restrict__ eo,
                                                 float* __restrict__ wpred) {
  extern __shared__ __align__(16) char smem[];
  __bf16* w1s = (__bf16*)smem;                 // [128][256] bf16 = 64KB (K-chunk of W1t[e])
  __bf16* w2s = w1s + H1 * 256;                // [64][128] bf16 = 16KB
  float*  w3s = (float*)(w2s + H2 * H1);       // 64
  float*  b1s = w3s + H2;                      // 128
  float*  b2s = b1s + H1;                      // 64
  __bf16* h1s = (__bf16*)(b2s + H2);           // 8 waves * [16][128] bf16 = 32KB

  const int lane = threadIdx.x & 31, w = threadIdx.x >> 5;
  const int lid = lane & 15, hi = lane >> 4;
  const int gRow = blockIdx.x * 128 + w * 16;
  const __bf16* xrow = xb + (size_t)(gRow + lid) * DDIM + hi * 8;
  __bf16* hw = h1s + w * 16 * H1;

  float wsum[8] = {0.f, 0.f, 0.f, 0.f, 0.f, 0.f, 0.f, 0.f};

  for (int e = 0; e < EDIM; ++e) {
    __syncthreads();  // previous iteration's LDS consumers done
    // stage per-expert layer-2/3 params
    {
      const uint4* src = (const uint4*)(w2t + (size_t)e * H2 * H1);
      for (int i = threadIdx.x; i < H2 * H1 / 8; i += 256) ((uint4*)w2s)[i] = src[i];
    }
    for (int i = threadIdx.x; i < H2; i += 256) w3s[i] = eW3[e * H2 + i];
    for (int i = threadIdx.x; i < H1; i += 256) b1s[i] = b1e[e * H1 + i];
    for (int i = threadIdx.x; i < H2; i += 256) b2s[i] = eb2g[e * H2 + i];

    // layer 1: h1 = relu(x @ W1_eff + b1_eff), K tiled through LDS
    v8f acc[8] = {};
    for (int kc = 0; kc < 4; ++kc) {
      __syncthreads();
      {
        const uint4* src = (const uint4*)(w1t + (size_t)e * H1 * DDIM + kc * 256);
        for (int i = threadIdx.x; i < H1 * 32; i += 256) {
          int h = i >> 5, c = i & 31;
          ((uint4*)w1s)[i] = src[(size_t)h * (DDIM / 8) + c];
        }
        if (kc < 3) {  // gfx1250 prefetch of next weight chunk
          __builtin_prefetch(w1t + (size_t)e * H1 * DDIM + (kc + 1) * 256 +
                                 (size_t)threadIdx.x * 512, 0, 1);
        }
      }
      __syncthreads();
      const __bf16* xk = xrow + kc * 256;

      // two-group pipeline: group0 = nt 0..3, group1 = nt 4..7
      ABf a0, a1, bA[4], bB[4];
      loadFragA(a0, xk);
#pragma unroll
      for (int nt = 0; nt < 4; ++nt)
        loadFrag(bA[nt], w1s + (nt * 16 + lid) * 256 + hi * 8);
#pragma unroll
      for (int k8 = 0; k8 < 8; ++k8) {
        // group1 loads for this step (latency covered by group0 WMMAs)
#pragma unroll
        for (int nt = 0; nt < 4; ++nt)
          loadFrag(bB[nt], w1s + ((nt + 4) * 16 + lid) * 256 + k8 * 32 + hi * 8);
        if (k8 < 7) loadFragA(a1, xk + (k8 + 1) * 32);
#pragma unroll
        for (int nt = 0; nt < 4; ++nt)
          acc[nt] = __builtin_amdgcn_wmma_f32_16x16x32_bf16(
              false, a0.v, false, bA[nt].v, (short)0, acc[nt], false, false);
        if (k8 < 7) {
          // next step's group0 loads (latency covered by group1 WMMAs)
#pragma unroll
          for (int nt = 0; nt < 4; ++nt)
            loadFrag(bA[nt], w1s + (nt * 16 + lid) * 256 + (k8 + 1) * 32 + hi * 8);
        }
#pragma unroll
        for (int nt = 0; nt < 4; ++nt)
          acc[nt + 4] = __builtin_amdgcn_wmma_f32_16x16x32_bf16(
              false, a0.v, false, bB[nt].v, (short)0, acc[nt + 4], false, false);
        if (k8 < 7) a0 = a1;
      }
    }

    // bias + relu -> bf16 h1 in wave-private LDS (A operand of layer 2)
#pragma unroll
    for (int nt = 0; nt < 8; ++nt) {
      int n = nt * 16 + lid;
      float bn = b1s[n];
#pragma unroll
      for (int r = 0; r < 8; ++r)
        hw[(r + hi * 8) * H1 + n] = f2bf(fmaxf(acc[nt][r] + bn, 0.f));
    }

    // layer 2: h2 = relu(h1 @ W2 + b2); short loop, simple preload is fine
    v8f acc2[4] = {};
#pragma unroll
    for (int k2 = 0; k2 < 4; ++k2) {
      const __bf16* ap = hw + lid * H1 + k2 * 32 + hi * 8;
      ABf a;
      loadFragA(a, ap);
      ABf b[4];
#pragma unroll
      for (int nt = 0; nt < 4; ++nt)
        loadFrag(b[nt], w2s + (nt * 16 + lid) * H1 + k2 * 32 + hi * 8);
#pragma unroll
      for (int nt = 0; nt < 4; ++nt)
        acc2[nt] = __builtin_amdgcn_wmma_f32_16x16x32_bf16(
            false, a.v, false, b[nt].v, (short)0, acc2[nt], false, false);
    }

    // layer 3 + sigmoid + routing-weighted accumulation
    float b2v[4], w3v[4];
#pragma unroll
    for (int nt = 0; nt < 4; ++nt) {
      b2v[nt] = b2s[nt * 16 + lid];
      w3v[nt] = w3s[nt * 16 + lid];
    }
    float eb3e = eb3[e];
#pragma unroll
    for (int r = 0; r < 8; ++r) {
      float s = 0.f;
#pragma unroll
      for (int nt = 0; nt < 4; ++nt)
        s = fmaf(fmaxf(acc2[nt][r] + b2v[nt], 0.f), w3v[nt], s);
      s += __shfl_xor(s, 1);
      s += __shfl_xor(s, 2);
      s += __shfl_xor(s, 4);
      s += __shfl_xor(s, 8);
      if (lid == 0) {  // lanes 0 (rows 0..7) and 16 (rows 8..15)
        int row = gRow + r + hi * 8;
        float sig = 1.f / (1.f + __expf(-(s + eb3e)));
        eo[(size_t)row * EDIM + e] = sig;
        wsum[r] = fmaf(rw[(size_t)row * EDIM + e], sig, wsum[r]);
      }
    }
  }

  if (lid == 0) {
#pragma unroll
    for (int r = 0; r < 8; ++r) wpred[gRow + r + hi * 8] = wsum[r];
  }
}

// ---------------- Host launcher ----------------

extern "C" void kernel_launch(void* const* d_in, const int* in_sizes, int n_in,
                              void* d_out, int out_size, void* d_ws, size_t ws_size,
                              hipStream_t stream) {
  const float* x       = (const float*)d_in[0];
  const float* r_gamma = (const float*)d_in[1];
  const float* r_beta  = (const float*)d_in[2];
  const float* r_mean  = (const float*)d_in[3];
  const float* r_var   = (const float*)d_in[4];
  const float* rW1     = (const float*)d_in[5];
  const float* rb1     = (const float*)d_in[6];
  const float* rW2     = (const float*)d_in[7];
  const float* rb2     = (const float*)d_in[8];
  const float* rW3     = (const float*)d_in[9];
  const float* rb3     = (const float*)d_in[10];
  const float* e_gamma = (const float*)d_in[11];
  const float* e_beta  = (const float*)d_in[12];
  const float* e_mean  = (const float*)d_in[13];
  const float* e_var   = (const float*)d_in[14];
  const float* eW1     = (const float*)d_in[15];
  const float* eb1     = (const float*)d_in[16];
  const float* eW2     = (const float*)d_in[17];
  const float* eb2     = (const float*)d_in[18];
  const float* eW3     = (const float*)d_in[19];
  const float* eb3     = (const float*)d_in[20];
  (void)in_sizes; (void)n_in; (void)out_size; (void)ws_size;

  // workspace partition
  char* ws = (char*)d_ws;
  size_t o = 0;
  __bf16* xb   = (__bf16*)(ws + o); o += (size_t)BDIM * DDIM * 2;      // 128 MB
  __bf16* w1t  = (__bf16*)(ws + o); o += (size_t)EDIM * H1 * DDIM * 2; // 4 MB
  __bf16* w2t  = (__bf16*)(ws + o); o += (size_t)EDIM * H2 * H1 * 2;   // 256 KB
  __bf16* rw1t = (__bf16*)(ws + o); o += (size_t)R1 * DDIM * 2;        // 128 KB
  float*  b1e  = (float*)(ws + o);  o += (size_t)EDIM * H1 * 4;
  float*  rb1e = (float*)(ws + o);  o += (size_t)R1 * 4;

  float* wpred = (float*)d_out;
  float* rwout = wpred + BDIM;
  float* eo    = rwout + (size_t)BDIM * EDIM;

  // prep
  k_cvt_x  <<<(BDIM * DDIM) / (256 * 8), 256, 0, stream>>>(x, xb);
  k_fold_w1<<<(EDIM * H1 * DDIM) / 256, 256, 0, stream>>>(eW1, e_gamma, e_var, w1t);
  k_fold_rw1<<<(R1 * DDIM) / 256, 256, 0, stream>>>(rW1, r_gamma, r_var, rw1t);
  k_b1eff  <<<EDIM * H1, 256, 0, stream>>>(eW1, e_gamma, e_beta, e_mean, e_var, eb1, b1e);
  k_rb1eff <<<R1, 256, 0, stream>>>(rW1, r_gamma, r_beta, r_mean, r_var, rb1, rb1e);
  k_w2t    <<<(EDIM * H2 * H1) / 256, 256, 0, stream>>>(eW2, w2t);

  // router (dynamic LDS: rW1t + tail params + padded h1)
  size_t r_smem = (size_t)R1 * DDIM * 2 +
                  4 * (size_t)(R1 + R1 * R2 + R2 + R2 * EDIM + EDIM + 8 * 16 * 65);
  k_router<<<BDIM / 128, 256, r_smem, stream>>>(xb, rw1t, rb1e, rW2, rb2, rW3, rb3, rwout);

  // experts (dynamic LDS: W1 K-chunk + W2 + params + per-wave h1)
  size_t e_smem = (size_t)H1 * 256 * 2 + (size_t)H2 * H1 * 2 +
                  4 * (size_t)(H2 + H1 + H2) + (size_t)8 * 16 * H1 * 2;
  k_experts<<<BDIM / 128, 256, e_smem, stream>>>(xb, w1t, b1e, w2t, eW3, eb2, eb3,
                                                 rwout, eo, wpred);
}
